// MultiHeadAttention_6425271075233
// MI455X (gfx1250) — compile-verified, matrix operand order checked by reading, not run on hardware
//
#include <hip/hip_runtime.h>
#include <hip/hip_bf16.h>

// Shapes from the reference
#define B_ 2
#define S_ 2048
#define D_ 1024
#define H_ 16
#define DH_ 64
#define LN_EPS 1e-5f

typedef __attribute__((ext_vector_type(16))) _Float16 v16h;
typedef __attribute__((ext_vector_type(8)))  _Float16 v8h;
typedef __attribute__((ext_vector_type(8)))  float    v8f;

// ---------------------------------------------------------------------------
// Fragment loader: 16-bit A(16x32) / B(32x16) fragment from a row-major matrix
// (for B, "rows" are columns of the logical B, i.e. pass Bt[n][k]).
// CDNA5 layout (ISA 7.12.2): lane L -> row L%16; hi=L/16;
//   elements j=0..7  : K = j          + 8*hi
//   elements j=8..15 : K = 16+(j-8)   + 8*hi
// -> two contiguous 16-byte loads per lane (works for global or LDS pointers).
// ---------------------------------------------------------------------------
__device__ __forceinline__ v16h load_frag16(const _Float16* base, int ld,
                                            int row0, int k0, int lane) {
  int r  = row0 + (lane & 15);
  int hi = (lane >> 4) & 1;
  const _Float16* p0 = base + (size_t)r * ld + k0 + 8 * hi;
  v8h lo = *(const v8h*)p0;
  v8h hv = *(const v8h*)(p0 + 16);
  v16h out;
#pragma unroll
  for (int j = 0; j < 8; ++j) { out[j] = lo[j]; out[j + 8] = hv[j]; }
  return out;
}

__device__ __forceinline__ v8f wmma_f16(v16h a, v16h b, v8f c) {
  // (neg_a, A, neg_b, B, c_mod, C, reuse_a, reuse_b)
  return __builtin_amdgcn_wmma_f32_16x16x32_f16(false, a, false, b, (short)0, c,
                                                false, false);
}

// ---------------------------------------------------------------------------
// Elementwise f32 -> f16
// ---------------------------------------------------------------------------
__global__ void cvt_f32_f16_kernel(const float* __restrict__ in,
                                   _Float16* __restrict__ out, size_t n) {
  size_t i = (size_t)blockIdx.x * blockDim.x + threadIdx.x;
  if (i < n) out[i] = (_Float16)in[i];
}

// W [K=1024, N=1024] f32 -> Wt [N][K] f16
__global__ void transpose_w_kernel(const float* __restrict__ in,
                                   _Float16* __restrict__ out) {
  size_t i = (size_t)blockIdx.x * blockDim.x + threadIdx.x;
  if (i >= (size_t)D_ * D_) return;
  int n = (int)(i >> 10);
  int k = (int)(i & 1023);
  out[i] = (_Float16)in[(size_t)k * D_ + n];
}

// ---------------------------------------------------------------------------
// WMMA GEMM, 2x4 register blocking: each wave computes a 32x64 tile of
// C[M,N] = A[M,K](f16) * Bt[N,K](f16)^T + bias.
// Per k-step: 2 A-frag loads + 4 B-frag loads feed 8 WMMAs.
// mode 0: store f16 row-major (Q/K projections)
// mode 1: store f16 transposed per batch: Vt[b][n][s]  (b=m/2048, s=m%2048)
// mode 2: store f32 row-major with residual add (output projection)
// ---------------------------------------------------------------------------
__global__ void __launch_bounds__(256)
gemm16_wmma_kernel(const _Float16* __restrict__ A,
                   const _Float16* __restrict__ Bt,
                   const float* __restrict__ bias,
                   const float* __restrict__ resid,
                   _Float16* __restrict__ out16,
                   float* __restrict__ out32,
                   int M, int N, int K, int mode) {
  int lane = threadIdx.x & 31;
  int wave = threadIdx.x >> 5;
  int tile = blockIdx.x * (blockDim.x >> 5) + wave;
  int tilesN = N >> 6;               // 64-wide N blocks
  int m0 = (tile / tilesN) << 5;     // 32 rows
  int n0 = (tile % tilesN) << 6;     // 64 cols
  if (m0 >= M) return;

  v8f acc[2][4];
#pragma unroll
  for (int mi = 0; mi < 2; ++mi)
#pragma unroll
    for (int ni = 0; ni < 4; ++ni) acc[mi][ni] = (v8f){};

  for (int k0 = 0; k0 < K; k0 += 32) {
    v16h a0 = load_frag16(A, K, m0,      k0, lane);
    v16h a1 = load_frag16(A, K, m0 + 16, k0, lane);
#pragma unroll
    for (int ni = 0; ni < 4; ++ni) {
      v16h bfr = load_frag16(Bt, K, n0 + ni * 16, k0, lane);
      acc[0][ni] = wmma_f16(a0, bfr, acc[0][ni]);
      acc[1][ni] = wmma_f16(a1, bfr, acc[1][ni]);
    }
  }

  int ln16 = lane & 15;
  int hi   = (lane >> 4) & 1;
#pragma unroll
  for (int mi = 0; mi < 2; ++mi) {
#pragma unroll
    for (int ni = 0; ni < 4; ++ni) {
      int n = n0 + ni * 16 + ln16;
      float bn = bias ? bias[n] : 0.0f;
#pragma unroll
      for (int r = 0; r < 8; ++r) {
        int m = m0 + mi * 16 + r + 8 * hi;
        float v = acc[mi][ni][r] + bn;
        if (mode == 2) {
          v += resid[(size_t)m * N + n];
          out32[(size_t)m * N + n] = v;
        } else if (mode == 1) {
          int b = m >> 11;
          int s = m & (S_ - 1);
          out16[((size_t)b * N + n) * S_ + s] = (_Float16)v;
        } else {
          out16[(size_t)m * N + n] = (_Float16)v;
        }
      }
    }
  }
}

// ---------------------------------------------------------------------------
// Fused attention: scores -> mask/scale -> softmax -> probs write (once) ->
// attn @ V, one 256-thread workgroup (8 waves) per (b, h, 16-query block).
// Wave w owns score columns [w*256, (w+1)*256): 16 v8f accumulators.
// The attn tensor (537 MB) is written exactly once and never re-read.
// The 32 KB contiguous mask block is DMA'd to LDS with
// GLOBAL_LOAD_ASYNC_TO_LDS_B128 at kernel entry, overlapping the score WMMAs;
// consumed after s_wait_asynccnt + barrier.
// ---------------------------------------------------------------------------
__global__ void __launch_bounds__(256)
fused_attn_kernel(const _Float16* __restrict__ Qp,
                  const _Float16* __restrict__ Kp,
                  const _Float16* __restrict__ Vt,
                  const unsigned char* __restrict__ mask,
                  float* __restrict__ attn,
                  _Float16* __restrict__ ctx) {
  __shared__ alignas(16) _Float16      sProbs[8][16][256]; // 64 KB: per-wave probs
  __shared__ alignas(16) float         sRed[8][16];        // cross-wave reductions
  __shared__ alignas(16) float         sCtx[8][16][64];    // 32 KB: partial context
  __shared__ alignas(16) unsigned char sMask[16][2048];    // 32 KB: mask block

  int t    = threadIdx.x;
  int lane = t & 31;
  int wave = t >> 5;
  int ln16 = lane & 15;
  int hi   = (lane >> 4) & 1;

  int wg = blockIdx.x;
  int qb = wg & 127;        // query block (16 rows)
  int bh = wg >> 7;         // 0..31
  int b  = bh >> 4, h = bh & 15;

  // --- async DMA of the 16x2048-byte mask block (contiguous) into LDS ---
  {
    const unsigned char* mbase =
        mask + (size_t)b * S_ * S_ + (size_t)(qb * 16) * S_;
    unsigned ldsBase = (unsigned)(uintptr_t)(&sMask[0][0]);
#pragma unroll
    for (int it = 0; it < 8; ++it) {
      unsigned off = (unsigned)(t + it * 256) * 16u;   // 16B per lane per op
      unsigned ldsOff = ldsBase + off;
      unsigned long long ga = (unsigned long long)(uintptr_t)(mbase + off);
      asm volatile("global_load_async_to_lds_b128 %0, %1, off"
                   :: "v"(ldsOff), "v"(ga) : "memory");
    }
  }

  const _Float16* Qb_ = Qp + (size_t)b * S_ * D_ + h * DH_;
  const _Float16* Kb_ = Kp + (size_t)b * S_ * D_ + h * DH_;
  const _Float16* Vb_ = Vt + (size_t)b * D_ * S_ + (size_t)(h * DH_) * S_;

  // Q fragments for this block (Dh = 64 -> two 16x32 fragments), reused 16x.
  v16h aq0 = load_frag16(Qb_, D_, qb * 16, 0,  lane);
  v16h aq1 = load_frag16(Qb_, D_, qb * 16, 32, lane);

  // ---- scores: 16 k-tiles per wave (columns kbase..kbase+255) ----
  int kbase = wave * 256;
  v8f acc[16];
#pragma unroll
  for (int i = 0; i < 16; ++i) {
    v8f c = {};
    c = wmma_f16(aq0, load_frag16(Kb_, D_, kbase + i * 16, 0,  lane), c);
    c = wmma_f16(aq1, load_frag16(Kb_, D_, kbase + i * 16, 32, lane), c);
    acc[i] = c;
  }

  // mask DMA must be complete (own wave's ops + all other waves')
  asm volatile("s_wait_asynccnt 0x0" ::: "memory");
  __syncthreads();

  // ---- scale + mask (mask from LDS) ----
#pragma unroll
  for (int i = 0; i < 16; ++i) {
    int kcol = kbase + i * 16 + ln16;
#pragma unroll
    for (int r = 0; r < 8; ++r) {
      int row = r + 8 * hi;
      float v = acc[i][r] * 0.125f;                // 1/sqrt(64)
      if (sMask[row][kcol]) v = -1e9f;
      acc[i][r] = v;
    }
  }

  // ---- row max: lane-local over 16 tiles, then across 16 lanes, then waves ----
  float m[8];
#pragma unroll
  for (int r = 0; r < 8; ++r) {
    float mm = -3.4e38f;
#pragma unroll
    for (int i = 0; i < 16; ++i) mm = fmaxf(mm, acc[i][r]);
    m[r] = mm;
  }
#pragma unroll
  for (int xm = 1; xm < 16; xm <<= 1)
#pragma unroll
    for (int r = 0; r < 8; ++r) m[r] = fmaxf(m[r], __shfl_xor(m[r], xm, 32));
  if (ln16 == 0) {
#pragma unroll
    for (int r = 0; r < 8; ++r) sRed[wave][r + 8 * hi] = m[r];
  }
  __syncthreads();
#pragma unroll
  for (int r = 0; r < 8; ++r) {
    float mm = sRed[0][r + 8 * hi];
#pragma unroll
    for (int wv = 1; wv < 8; ++wv) mm = fmaxf(mm, sRed[wv][r + 8 * hi]);
    m[r] = mm;
  }
  __syncthreads();   // sRed reused for sums

  // ---- exp + row sum ----
  float s[8];
#pragma unroll
  for (int r = 0; r < 8; ++r) s[r] = 0.0f;
#pragma unroll
  for (int i = 0; i < 16; ++i)
#pragma unroll
    for (int r = 0; r < 8; ++r) {
      float e = __expf(acc[i][r] - m[r]);
      acc[i][r] = e;
      s[r] += e;
    }
#pragma unroll
  for (int xm = 1; xm < 16; xm <<= 1)
#pragma unroll
    for (int r = 0; r < 8; ++r) s[r] += __shfl_xor(s[r], xm, 32);
  if (ln16 == 0) {
#pragma unroll
    for (int r = 0; r < 8; ++r) sRed[wave][r + 8 * hi] = s[r];
  }
  __syncthreads();
#pragma unroll
  for (int r = 0; r < 8; ++r) {
    float ss = 0.0f;
#pragma unroll
    for (int wv = 0; wv < 8; ++wv) ss += sRed[wv][r + 8 * hi];
    s[r] = 1.0f / ss;
  }

  // ---- normalize; write probs to d_out (once) and stage f16 in LDS ----
  float* arow = attn + (size_t)bh * S_ * S_ + (size_t)(qb * 16) * S_;
#pragma unroll
  for (int i = 0; i < 16; ++i)
#pragma unroll
    for (int r = 0; r < 8; ++r) {
      float p = acc[i][r] * s[r];
      int row = r + 8 * hi;
      arow[(size_t)row * S_ + kbase + i * 16 + ln16] = p;
      sProbs[wave][row][i * 16 + ln16] = (_Float16)p;
    }
  __syncthreads();

  // ---- context: probs (LDS, A-frags) x V (global, B-frags from Vt[b][d][s]) ----
  v8f cacc[4];
#pragma unroll
  for (int dt = 0; dt < 4; ++dt) cacc[dt] = (v8f){};
  const _Float16* myProbs = &sProbs[wave][0][0];   // 16 x 256, ld = 256
#pragma unroll
  for (int j = 0; j < 8; ++j) {
    v16h ap = load_frag16(myProbs, 256, 0, j * 32, lane);
#pragma unroll
    for (int dt = 0; dt < 4; ++dt) {
      v16h bv = load_frag16(Vb_ + (size_t)dt * 16 * S_, S_, 0,
                            kbase + j * 32, lane);
      cacc[dt] = wmma_f16(ap, bv, cacc[dt]);
    }
  }
  // partial contexts -> LDS
#pragma unroll
  for (int dt = 0; dt < 4; ++dt)
#pragma unroll
    for (int r = 0; r < 8; ++r)
      sCtx[wave][r + 8 * hi][dt * 16 + ln16] = cacc[dt][r];
  __syncthreads();

  // reduce 8 wave-partials; store ctx f16 in flat (b,h,s,dh) order (the
  // reference's buggy reshape).
  for (int e = t; e < 16 * 64; e += 256) {
    int row = e >> 6, col = e & 63;
    float sum = 0.0f;
#pragma unroll
    for (int wv = 0; wv < 8; ++wv) sum += sCtx[wv][row][col];
    ctx[((size_t)bh * S_ + qb * 16 + row) * DH_ + col] = (_Float16)sum;
  }
}

// ---------------------------------------------------------------------------
// LayerNorm over D=1024; one 256-thread block per row.
// ---------------------------------------------------------------------------
__global__ void ln_kernel(const float* __restrict__ x,
                          const float* __restrict__ gamma,
                          const float* __restrict__ beta,
                          float* __restrict__ out) {
  __shared__ float red[256];
  const float* p = x + (size_t)blockIdx.x * D_;
  int t = threadIdx.x;
  float v[4];
  float s = 0.0f;
#pragma unroll
  for (int i = 0; i < 4; ++i) { v[i] = p[t + 256 * i]; s += v[i]; }
  red[t] = s; __syncthreads();
  for (int st = 128; st > 0; st >>= 1) { if (t < st) red[t] += red[t + st]; __syncthreads(); }
  float mu = red[0] * (1.0f / D_); __syncthreads();
  float vs = 0.0f;
#pragma unroll
  for (int i = 0; i < 4; ++i) { float d = v[i] - mu; vs += d * d; }
  red[t] = vs; __syncthreads();
  for (int st = 128; st > 0; st >>= 1) { if (t < st) red[t] += red[t + st]; __syncthreads(); }
  float inv = rsqrtf(red[0] * (1.0f / D_) + LN_EPS);
  float* o = out + (size_t)blockIdx.x * D_;
#pragma unroll
  for (int i = 0; i < 4; ++i) {
    int c = t + 256 * i;
    o[c] = (v[i] - mu) * inv * gamma[c] + beta[c];
  }
}

// ---------------------------------------------------------------------------
extern "C" void kernel_launch(void* const* d_in, const int* in_sizes, int n_in,
                              void* d_out, int out_size, void* d_ws, size_t ws_size,
                              hipStream_t stream) {
  (void)in_sizes; (void)n_in; (void)out_size; (void)ws_size;
  const float* q    = (const float*)d_in[0];
  const float* k    = (const float*)d_in[1];
  const float* v    = (const float*)d_in[2];
  const unsigned char* mask = (const unsigned char*)d_in[3]; // jax bool = 1 byte
  const float* Wq = (const float*)d_in[4];  const float* bq = (const float*)d_in[5];
  const float* Wk = (const float*)d_in[6];  const float* bk = (const float*)d_in[7];
  const float* Wv = (const float*)d_in[8];  const float* bv = (const float*)d_in[9];
  const float* Wo = (const float*)d_in[10]; const float* bo = (const float*)d_in[11];
  const float* gamma = (const float*)d_in[12];
  const float* beta  = (const float*)d_in[13];

  float* out  = (float*)d_out;                         // [B,S,D]
  float* attn = out + (size_t)B_ * S_ * D_;            // [B,H,S,S]

  char* w = (char*)d_ws;
  const size_t MB = (size_t)1 << 20;
  _Float16* q16 = (_Float16*)(w + 0 * MB);    // 8 MB each (4096x1024 f16)
  _Float16* k16 = (_Float16*)(w + 8 * MB);
  _Float16* v16 = (_Float16*)(w + 16 * MB);
  _Float16* Wqt = (_Float16*)(w + 24 * MB);   // 2 MB each (1024x1024 f16)
  _Float16* Wkt = (_Float16*)(w + 26 * MB);
  _Float16* Wvt = (_Float16*)(w + 28 * MB);
  _Float16* Wot = (_Float16*)(w + 30 * MB);
  _Float16* Qp  = (_Float16*)(w + 32 * MB);   // projected Q  [B*S, D] f16
  _Float16* Kp  = (_Float16*)(w + 40 * MB);   // projected K  [B*S, D] f16
  _Float16* Vt  = (_Float16*)(w + 48 * MB);   // projected V, transposed [B][D][S] f16
  _Float16* ctx = (_Float16*)(w + 56 * MB);   // context flat (b,h,s,dh) f16
  float*    xbf = (float*)   (w + 64 * MB);   // pre-LN activations f32 (16 MB)

  const size_t nAct = (size_t)B_ * S_ * D_;   // 4194304
  const int CVT_B = 256;
  cvt_f32_f16_kernel<<<(unsigned)((nAct + CVT_B - 1) / CVT_B), CVT_B, 0, stream>>>(q, q16, nAct);
  cvt_f32_f16_kernel<<<(unsigned)((nAct + CVT_B - 1) / CVT_B), CVT_B, 0, stream>>>(k, k16, nAct);
  cvt_f32_f16_kernel<<<(unsigned)((nAct + CVT_B - 1) / CVT_B), CVT_B, 0, stream>>>(v, v16, nAct);

  const unsigned wBlocks = (unsigned)(((size_t)D_ * D_) / CVT_B);
  transpose_w_kernel<<<wBlocks, CVT_B, 0, stream>>>(Wq, Wqt);
  transpose_w_kernel<<<wBlocks, CVT_B, 0, stream>>>(Wk, Wkt);
  transpose_w_kernel<<<wBlocks, CVT_B, 0, stream>>>(Wv, Wvt);
  transpose_w_kernel<<<wBlocks, CVT_B, 0, stream>>>(Wo, Wot);

  // Projection GEMMs: M=4096, N=1024, K=1024; 32x64 per wave ->
  // (4096/32)*(1024/64) = 2048 waves -> 256 blocks of 8 waves.
  const int M = B_ * S_, N = D_, K = D_;
  const unsigned gemmBlocks = (unsigned)((M / 32) * (N / 64) / 8);
  gemm16_wmma_kernel<<<gemmBlocks, 256, 0, stream>>>(q16, Wqt, bq, nullptr, Qp, nullptr, M, N, K, 0);
  gemm16_wmma_kernel<<<gemmBlocks, 256, 0, stream>>>(k16, Wkt, bk, nullptr, Kp, nullptr, M, N, K, 0);
  gemm16_wmma_kernel<<<gemmBlocks, 256, 0, stream>>>(v16, Wvt, bv, nullptr, Vt, nullptr, M, N, K, 1);

  // Fused scores+softmax+AV: one WG per (bh, 16-q block): 32*128 = 4096 WGs.
  fused_attn_kernel<<<4096, 256, 0, stream>>>(Qp, Kp, Vt, mask, attn, ctx);

  // Output projection + residual -> xbf, then LayerNorm -> out.
  gemm16_wmma_kernel<<<gemmBlocks, 256, 0, stream>>>(ctx, Wot, bo, q, nullptr, xbf, M, N, K, 2);
  ln_kernel<<<(unsigned)M, 256, 0, stream>>>(xbf, gamma, beta, out);
}
